// EdgeAwareAttention_24283745091779
// MI455X (gfx1250) — compile-verified, matrix operand order checked by reading, not run on hardware
//
#include <hip/hip_runtime.h>
#include <stdint.h>

// Problem constants (reference: B,N,D,H,E = 4,1024,512,8,5; HD=64)
#define BB 4
#define NN 1024
#define DD 512
#define HH 8
#define EE 5
#define HD 64
#define BIG_NEG (-3.0e38f)

typedef __attribute__((ext_vector_type(16))) _Float16 v16h;
typedef __attribute__((ext_vector_type(8)))  float    v8f;

// ---- gfx1250 async-to-LDS path (guarded so both toolchains compile) -------
#if defined(__has_builtin)
#if __has_builtin(__builtin_amdgcn_global_load_async_to_lds_b32)
#define HAVE_ASYNC_LDS 1
#endif
#if __has_builtin(__builtin_amdgcn_s_wait_asynccnt)
#define HAVE_WAIT_ASYNC 1
#endif
#endif

typedef __attribute__((address_space(1))) int gas_int;   // global
typedef __attribute__((address_space(3))) int las_int;   // LDS

__device__ __forceinline__ void wait_asynccnt0() {
#if defined(HAVE_WAIT_ASYNC)
  __builtin_amdgcn_s_wait_asynccnt(0);
#elif defined(HAVE_ASYNC_LDS)
  asm volatile("s_wait_asynccnt 0" ::: "memory");
#endif
}

// ---------------------------------------------------------------------------
// Fragment loaders. A-operand (16 rows x 32 K, f16): lanes 0-15 hold rows 0-15
// (K = 0..7,16..23), lanes 16-31 hold the same rows (K = 8..15,24..31).
// half j -> K = (j<8 ? j : j+8) + 8*hi.
// ---------------------------------------------------------------------------
__device__ __forceinline__ v16h load_frag_rowmajor(const _Float16* __restrict__ p,
                                                   int ld, int lane) {
  int row = lane & 15, hi = lane >> 4;
  v16h r;
#pragma unroll
  for (int j = 0; j < 16; ++j) {
    int K = (j < 8 ? j : j + 8) + hi * 8;
    r[j] = p[(size_t)row * ld + K];
  }
  return r;
}

// B-operand (32 K x 16 cols): col = lane&15, K strided by strideK elements.
__device__ __forceinline__ v16h load_frag_b_strided(const _Float16* __restrict__ p,
                                                    int strideK, int lane) {
  int n = lane & 15, hi = lane >> 4;
  v16h r;
#pragma unroll
  for (int j = 0; j < 16; ++j) {
    int K = (j < 8 ? j : j + 8) + hi * 8;
    r[j] = p[(size_t)K * strideK + n];
  }
  return r;
}

// ---------------------------------------------------------------------------
// Kernel 1: fp32 -> fp16 conversion (grid-stride)
// ---------------------------------------------------------------------------
__global__ void eaa_f32_to_f16(const float* __restrict__ s, _Float16* __restrict__ d, int n) {
  for (int i = blockIdx.x * blockDim.x + threadIdx.x; i < n; i += gridDim.x * blockDim.x)
    d[i] = (_Float16)s[i];
}

// ---------------------------------------------------------------------------
// Kernel 2: QKV GEMM.  C[4096x1536] = Xh[4096x512] * Wqkv^T  (w row-major [c][k])
// Each wave: one 16x64 output tile via WMMA; scatter-store into q / kT / v.
// ---------------------------------------------------------------------------
__global__ void eaa_qkv_gemm(const _Float16* __restrict__ xh,
                             const _Float16* __restrict__ wh,
                             _Float16* __restrict__ qh,
                             _Float16* __restrict__ kTh,
                             _Float16* __restrict__ vh) {
  int wid  = (blockIdx.x * blockDim.x + threadIdx.x) >> 5;
  int lane = threadIdx.x & 31;
  int mt = wid / 24;        // 256 row tiles
  int ct = wid % 24;        // 24 col tiles of 64
  v8f acc[4] = {v8f{}, v8f{}, v8f{}, v8f{}};
  const _Float16* arow = xh + (size_t)(mt * 16) * DD;
  const _Float16* wrow = wh + (size_t)(ct * 64) * DD;
  for (int kb = 0; kb < DD; kb += 32) {
    if (kb + 32 < DD) {   // prefetch next k-step tiles into near caches
      __builtin_prefetch(arow + kb + 32, 0, 3);
      __builtin_prefetch(wrow + kb + 32, 0, 3);
    }
    // issue all 5 fragment loads, then 4 back-to-back WMMAs
    v16h a  = load_frag_rowmajor(arow + kb, DD, lane);
    v16h b0 = load_frag_rowmajor(wrow + (size_t)( 0) * DD + kb, DD, lane);
    v16h b1 = load_frag_rowmajor(wrow + (size_t)(16) * DD + kb, DD, lane);
    v16h b2 = load_frag_rowmajor(wrow + (size_t)(32) * DD + kb, DD, lane);
    v16h b3 = load_frag_rowmajor(wrow + (size_t)(48) * DD + kb, DD, lane);
    acc[0] = __builtin_amdgcn_wmma_f32_16x16x32_f16(false, a, false, b0, (short)0, acc[0], false, false);
    acc[1] = __builtin_amdgcn_wmma_f32_16x16x32_f16(false, a, false, b1, (short)0, acc[1], false, false);
    acc[2] = __builtin_amdgcn_wmma_f32_16x16x32_f16(false, a, false, b2, (short)0, acc[2], false, false);
    acc[3] = __builtin_amdgcn_wmma_f32_16x16x32_f16(false, a, false, b3, (short)0, acc[3], false, false);
  }
  int hi = lane >> 4, lo = lane & 15;
#pragma unroll
  for (int t = 0; t < 4; ++t)
#pragma unroll
    for (int r = 0; r < 8; ++r) {
      int row = mt * 16 + r + 8 * hi;        // 0..4095
      int col = ct * 64 + t * 16 + lo;       // 0..1535
      int b   = row >> 10, nn = row & (NN - 1);
      int sec = col >> 9;                    // 0=q 1=k 2=v
      int cc  = col & (DD - 1);
      int h   = cc >> 6, dd = cc & (HD - 1);
      _Float16 hv = (_Float16)acc[t][r];
      size_t bh = (size_t)(b * HH + h);
      if (sec == 0)      qh [(bh * NN + nn) * HD + dd] = hv;
      else if (sec == 1) kTh[(bh * HD + dd) * NN + nn] = hv;   // transposed
      else               vh [(bh * NN + nn) * HD + dd] = hv;
    }
}

// ---------------------------------------------------------------------------
// Kernel 3: fused edge-aware flash attention.
// grid = B * N/16 blocks; block = 256 threads = 8 waves; wave w == head w.
// Edge tile (16 q-rows x 32 cols x 5) staged to LDS ONCE per chunk (async-to-
// LDS, overlapped with the S matmul), shared by all 8 heads.
// ---------------------------------------------------------------------------
__global__ void eaa_attention(const _Float16* __restrict__ qh,
                              const _Float16* __restrict__ kTh,
                              const _Float16* __restrict__ vh,
                              const float* __restrict__ ef,
                              const unsigned char* __restrict__ nmask,
                              const float* __restrict__ w_ep,
                              const float* __restrict__ w_eg,
                              const float* __restrict__ b_eg,
                              _Float16* __restrict__ ao) {
  __shared__ float    efs[16 * 32 * EE];        // 10 KB edge tile
  __shared__ _Float16 pst[HH][16 * 32];         // per-wave P re-layout buffer

  const int b   = blockIdx.x >> 6;              // N/16 = 64 q-tiles per batch
  const int q0  = (blockIdx.x & 63) * 16;
  const int w   = threadIdx.x >> 5;             // wave id == head
  const int lane = threadIdx.x & 31;
  const int hi = lane >> 4, lo = lane & 15;
  const float scale = 0.125f;                   // 1/sqrt(64)

  float wep[EE], weg[EE];
#pragma unroll
  for (int e = 0; e < EE; ++e) { wep[e] = w_ep[w * EE + e]; weg[e] = w_eg[w * EE + e]; }
  const float beg = b_eg[w];

  // Q fragments for K-dim 0..31 and 32..63
  const _Float16* qbase = qh + ((size_t)(b * HH + w) * NN + q0) * HD;
  v16h a0 = load_frag_rowmajor(qbase,      HD, lane);
  v16h a1 = load_frag_rowmajor(qbase + 32, HD, lane);

  unsigned qm[8];
#pragma unroll
  for (int r = 0; r < 8; ++r) qm[r] = nmask[b * NN + q0 + r + 8 * hi];

  v8f o[4] = {v8f{}, v8f{}, v8f{}, v8f{}};
  float mrow[8], lrow[8];
#pragma unroll
  for (int r = 0; r < 8; ++r) { mrow[r] = BIG_NEG; lrow[r] = 0.0f; }

  const _Float16* kbase = kTh + (size_t)(b * HH + w) * HD * NN;
  const _Float16* vbase = vh  + (size_t)(b * HH + w) * NN * HD;

  for (int mc = 0; mc < NN; mc += 32) {
    // ---- stage edge tile: 16 rows x 160 contiguous floats ------------------
    __syncthreads();
    {
      const float* src = ef + ((size_t)(b * NN + q0) * NN + mc) * EE;
#pragma unroll
      for (int i = 0; i < 10; ++i) {
        int idx = threadIdx.x * 10 + i;                 // 0..2559
        int qr = idx / 160, rem = idx - qr * 160;
        const float* gp = src + (size_t)qr * NN * EE + rem;
        float* lp = &efs[qr * 160 + rem];
#if defined(HAVE_ASYNC_LDS)
        __builtin_amdgcn_global_load_async_to_lds_b32((gas_int*)gp, (las_int*)lp, 0, 0);
#else
        *lp = *gp;
#endif
      }
    }
    // ---- S = Q K^T (16x32): 4 B-frag loads, then 4 WMMAs -------------------
    v8f s0 = v8f{}, s1 = v8f{};
    {
      v16h bk0 = load_frag_b_strided(kbase + mc,                NN, lane);
      v16h bk1 = load_frag_b_strided(kbase + mc + 16,           NN, lane);
      v16h bk2 = load_frag_b_strided(kbase + 32 * NN + mc,      NN, lane);
      v16h bk3 = load_frag_b_strided(kbase + 32 * NN + mc + 16, NN, lane);
      s0 = __builtin_amdgcn_wmma_f32_16x16x32_f16(false, a0, false, bk0, (short)0, s0, false, false);
      s1 = __builtin_amdgcn_wmma_f32_16x16x32_f16(false, a0, false, bk1, (short)0, s1, false, false);
      s0 = __builtin_amdgcn_wmma_f32_16x16x32_f16(false, a1, false, bk2, (short)0, s0, false, false);
      s1 = __builtin_amdgcn_wmma_f32_16x16x32_f16(false, a1, false, bk3, (short)0, s1, false, false);
    }
    wait_asynccnt0();   // edge tile landed in LDS (overlapped with S matmul)
    __syncthreads();

    // ---- edge bias + sigmoid gate + mask (C layout: row=r+8*hi, col=lo) ----
    unsigned cm0 = nmask[b * NN + mc + lo];
    unsigned cm1 = nmask[b * NN + mc + 16 + lo];
#pragma unroll
    for (int r = 0; r < 8; ++r) {
      int rr = r + 8 * hi;
      const float* e0 = &efs[rr * 160 + lo * EE];
      const float* e1 = &efs[rr * 160 + (lo + 16) * EE];
      float bias0 = 0.f, gz0 = 0.f, bias1 = 0.f, gz1 = 0.f;
#pragma unroll
      for (int e = 0; e < EE; ++e) {
        bias0 += e0[e] * wep[e]; gz0 += e0[e] * weg[e];
        bias1 += e1[e] * wep[e]; gz1 += e1[e] * weg[e];
      }
      float g0 = 1.0f / (1.0f + __expf(-(gz0 + beg)));
      float g1 = 1.0f / (1.0f + __expf(-(gz1 + beg)));
      float v0 = s0[r] * scale + g0 * bias0;
      float v1 = s1[r] * scale + g1 * bias1;
      s0[r] = (qm[r] && cm0) ? v0 : BIG_NEG;
      s1[r] = (qm[r] && cm1) ? v1 : BIG_NEG;
    }

    // ---- online softmax: row max/sum across the 16 lanes of each half ------
    float rmax[8];
#pragma unroll
    for (int r = 0; r < 8; ++r) rmax[r] = fmaxf(s0[r], s1[r]);
#pragma unroll
    for (int m = 1; m <= 8; m <<= 1)
#pragma unroll
      for (int r = 0; r < 8; ++r) rmax[r] = fmaxf(rmax[r], __shfl_xor(rmax[r], m, 32));
    float rsum[8];
#pragma unroll
    for (int r = 0; r < 8; ++r) {
      float nmx = fmaxf(mrow[r], rmax[r]);
      float alpha = __expf(mrow[r] - nmx);
      mrow[r] = nmx;
      float p0 = __expf(s0[r] - nmx);
      float p1 = __expf(s1[r] - nmx);
      s0[r] = p0; s1[r] = p1;
      rsum[r] = p0 + p1;
      lrow[r] = lrow[r] * alpha;
#pragma unroll
      for (int t = 0; t < 4; ++t) o[t][r] *= alpha;
    }
#pragma unroll
    for (int m = 1; m <= 8; m <<= 1)
#pragma unroll
      for (int r = 0; r < 8; ++r) rsum[r] += __shfl_xor(rsum[r], m, 32);
#pragma unroll
    for (int r = 0; r < 8; ++r) lrow[r] += rsum[r];

    // ---- re-layout P (C layout -> A layout) through LDS --------------------
#pragma unroll
    for (int r = 0; r < 8; ++r) {
      pst[w][(r + 8 * hi) * 32 + lo]      = (_Float16)s0[r];
      pst[w][(r + 8 * hi) * 32 + lo + 16] = (_Float16)s1[r];
    }
    asm volatile("s_wait_dscnt 0" ::: "memory");
    v16h ap;
    {
      int row = lane & 15;
#pragma unroll
      for (int j = 0; j < 16; ++j) {
        int K = (j < 8 ? j : j + 8) + hi * 8;
        ap[j] = pst[w][row * 32 + K];
      }
    }

    // ---- O += P * V  (4 n-tiles of 16 over HD=64) --------------------------
    v16h bv0 = load_frag_b_strided(vbase + (size_t)mc * HD +  0, HD, lane);
    v16h bv1 = load_frag_b_strided(vbase + (size_t)mc * HD + 16, HD, lane);
    v16h bv2 = load_frag_b_strided(vbase + (size_t)mc * HD + 32, HD, lane);
    v16h bv3 = load_frag_b_strided(vbase + (size_t)mc * HD + 48, HD, lane);
    o[0] = __builtin_amdgcn_wmma_f32_16x16x32_f16(false, ap, false, bv0, (short)0, o[0], false, false);
    o[1] = __builtin_amdgcn_wmma_f32_16x16x32_f16(false, ap, false, bv1, (short)0, o[1], false, false);
    o[2] = __builtin_amdgcn_wmma_f32_16x16x32_f16(false, ap, false, bv2, (short)0, o[2], false, false);
    o[3] = __builtin_amdgcn_wmma_f32_16x16x32_f16(false, ap, false, bv3, (short)0, o[3], false, false);
  }

  // ---- normalize and store (layout (b, n, h*64+d) = (b,n,D) for proj GEMM) -
  float invl[8];
#pragma unroll
  for (int r = 0; r < 8; ++r) invl[r] = 1.0f / lrow[r];
#pragma unroll
  for (int t = 0; t < 4; ++t)
#pragma unroll
    for (int r = 0; r < 8; ++r) {
      int n = q0 + r + 8 * hi;
      int col = w * HD + t * 16 + lo;
      ao[(size_t)(b * NN + n) * DD + col] = (_Float16)(o[t][r] * invl[r]);
    }
}

// ---------------------------------------------------------------------------
// Kernel 4: output projection.  out[4096x512] = AO[4096x512] * Wproj^T + bias
// ---------------------------------------------------------------------------
__global__ void eaa_proj_gemm(const _Float16* __restrict__ aoh,
                              const _Float16* __restrict__ wh,
                              const float* __restrict__ bias,
                              float* __restrict__ out) {
  int wid  = (blockIdx.x * blockDim.x + threadIdx.x) >> 5;
  int lane = threadIdx.x & 31;
  int mt = wid >> 3;            // 256 row tiles
  int ct = wid & 7;             // 8 col tiles of 64
  v8f acc[4] = {v8f{}, v8f{}, v8f{}, v8f{}};
  const _Float16* arow = aoh + (size_t)(mt * 16) * DD;
  const _Float16* wrow = wh + (size_t)(ct * 64) * DD;
  for (int kb = 0; kb < DD; kb += 32) {
    if (kb + 32 < DD) {
      __builtin_prefetch(arow + kb + 32, 0, 3);
      __builtin_prefetch(wrow + kb + 32, 0, 3);
    }
    v16h a  = load_frag_rowmajor(arow + kb, DD, lane);
    v16h b0 = load_frag_rowmajor(wrow + (size_t)( 0) * DD + kb, DD, lane);
    v16h b1 = load_frag_rowmajor(wrow + (size_t)(16) * DD + kb, DD, lane);
    v16h b2 = load_frag_rowmajor(wrow + (size_t)(32) * DD + kb, DD, lane);
    v16h b3 = load_frag_rowmajor(wrow + (size_t)(48) * DD + kb, DD, lane);
    acc[0] = __builtin_amdgcn_wmma_f32_16x16x32_f16(false, a, false, b0, (short)0, acc[0], false, false);
    acc[1] = __builtin_amdgcn_wmma_f32_16x16x32_f16(false, a, false, b1, (short)0, acc[1], false, false);
    acc[2] = __builtin_amdgcn_wmma_f32_16x16x32_f16(false, a, false, b2, (short)0, acc[2], false, false);
    acc[3] = __builtin_amdgcn_wmma_f32_16x16x32_f16(false, a, false, b3, (short)0, acc[3], false, false);
  }
  int hi = lane >> 4, lo = lane & 15;
#pragma unroll
  for (int t = 0; t < 4; ++t)
#pragma unroll
    for (int r = 0; r < 8; ++r) {
      int row = mt * 16 + r + 8 * hi;
      int col = ct * 64 + t * 16 + lo;
      out[(size_t)row * DD + col] = acc[t][r] + bias[col];
    }
}

// ---------------------------------------------------------------------------
extern "C" void kernel_launch(void* const* d_in, const int* in_sizes, int n_in,
                              void* d_out, int out_size, void* d_ws, size_t ws_size,
                              hipStream_t stream) {
  const float* x      = (const float*)d_in[0];
  const float* ef     = (const float*)d_in[1];
  const unsigned char* nmask = (const unsigned char*)d_in[2]; // jax bool = 1 byte
  const float* w_qkv  = (const float*)d_in[3];
  const float* w_ep   = (const float*)d_in[4];
  const float* w_eg   = (const float*)d_in[5];
  const float* b_eg   = (const float*)d_in[6];
  const float* w_proj = (const float*)d_in[7];
  const float* b_proj = (const float*)d_in[8];
  float* out = (float*)d_out;

  // workspace carve-up (f16 buffers), 256B aligned
  char* ws = (char*)d_ws;
  size_t off = 0;
  auto carve = [&](size_t bytes) {
    char* p = ws + off;
    off = (off + bytes + 255) & ~(size_t)255;
    return p;
  };
  _Float16* xh     = (_Float16*)carve((size_t)BB * NN * DD * 2);        // 4 MB
  _Float16* wqkvh  = (_Float16*)carve((size_t)3 * DD * DD * 2);         // 1.5 MB
  _Float16* wprojh = (_Float16*)carve((size_t)DD * DD * 2);             // 0.5 MB
  _Float16* qh     = (_Float16*)carve((size_t)BB * HH * NN * HD * 2);   // 4 MB
  _Float16* kTh    = (_Float16*)carve((size_t)BB * HH * HD * NN * 2);   // 4 MB
  _Float16* vh     = (_Float16*)carve((size_t)BB * HH * NN * HD * 2);   // 4 MB
  _Float16* ao     = (_Float16*)carve((size_t)BB * NN * DD * 2);        // 4 MB
  (void)ws_size; (void)in_sizes; (void)n_in; (void)out_size;

  eaa_f32_to_f16<<<2048, 256, 0, stream>>>(x,      xh,     BB * NN * DD);
  eaa_f32_to_f16<<<1024, 256, 0, stream>>>(w_qkv,  wqkvh,  3 * DD * DD);
  eaa_f32_to_f16<<<512,  256, 0, stream>>>(w_proj, wprojh, DD * DD);

  // 4096/16 * 1536/64 = 6144 waves -> 768 blocks of 8 waves
  eaa_qkv_gemm<<<768, 256, 0, stream>>>(xh, wqkvh, qh, kTh, vh);

  // B * N/16 = 256 blocks; 8 waves/block = 8 heads
  eaa_attention<<<256, 256, 0, stream>>>(qh, kTh, vh, ef, nmask,
                                         w_ep, w_eg, b_eg, ao);

  // 4096/16 * 512/64 = 2048 waves -> 256 blocks
  eaa_proj_gemm<<<256, 256, 0, stream>>>(ao, wprojh, b_proj, out);
}
